// SSAB_14096082665926
// MI455X (gfx1250) — compile-verified, matrix operand order checked by reading, not run on hardware
//
#include <hip/hip_runtime.h>
#include <hip/hip_bf16.h>
#include <math.h>

// ---------------------------------------------------------------------------
// SSAB block for MI455X (gfx1250).
//  * All 1x1 convs + the q/k Gram matrix use v_wmma_f32_16x16x32_f16
//    (f16 in, f32 accum, K=64 -> negligible rounding).
//  * Activation/weight tiles reach LDS via the Tensor Data Mover
//    (tensor_load_to_lds + s_wait_tensorcnt), with D# LDS padding
//    (row stride 68 dwords) to avoid LDS bank conflicts.
//  * Workload is memory bound (~1 GB traffic @ 23.3 TB/s vs ~19 GFLOP),
//    so the restructured GEMM reads each activation tile from HBM once
//    and loops cout tiles out of LDS.
// ---------------------------------------------------------------------------

typedef __attribute__((ext_vector_type(16))) _Float16 v16h;
typedef __attribute__((ext_vector_type(8)))  float    v8f;
typedef __attribute__((ext_vector_type(4)))  unsigned v4u;
typedef __attribute__((ext_vector_type(8)))  unsigned v8u;

#define DIM    64
#define HEADS  8
#define NB     2
#define BATCH  2
#define IMH    256
#define IMW    256
#define NPIX   (IMH * IMW)          // 65536
#define EPS_LN 1e-6f

#define LDS_ROW 68                  // 64 dwords + 4 dwords TDM pad
#define W_ROWS_MAX 128              // cout rows per chunk staged in LDS
#define PW_SHMEM ((size_t)(W_ROWS_MAX + 64) * LDS_ROW * sizeof(float))

// ---------------------------------------------------------------------------
// Issue one TDM 2D tile load (f32 elements): rows x cols tile, source row
// stride rowStride (dwords), into LDS at ldsPtr with 4-dword padding every
// 64 dwords (so a 64-wide row lands at stride 68 in LDS).
// D# layout per CDNA5 ISA sec.8.
// ---------------------------------------------------------------------------
__device__ __forceinline__ void tdm_load_tile_f32(
    const float* gptr, float* ldsPtr, unsigned rows, unsigned cols,
    unsigned long rowStride)
{
  const unsigned long ga  = (unsigned long)gptr;
  const unsigned      lds = (unsigned)(unsigned long)ldsPtr;
  const v4u g0 = { 0x1u,                                   // count=1, user D#
                   lds,                                    // lds_addr
                   (unsigned)ga,                           // global_addr[31:0]
                   (unsigned)((ga >> 32) & 0x01FFFFFFu)    // global_addr[56:32]
                       | (2u << 30) };                     // type=2 ("image")
  const unsigned w0 = (2u << 16)     // data_size = 4 bytes
                    | (1u << 20)     // pad_enable
                    | (5u << 22)     // pad_interval: 64 dwords
                    | (3u << 25);    // pad_amount:   4 dwords
  const v8u g1 = { w0,
                   (cols & 0xFFFFu) << 16,                       // tensor_dim0 lo
                   (cols >> 16) | ((rows & 0xFFFFu) << 16),      // d0 hi | tensor_dim1 lo
                   (rows >> 16) | ((cols & 0xFFFFu) << 16),      // d1 hi | tile_dim0
                   rows & 0xFFFFu,                               // tile_dim1 (tile_dim2=0)
                   (unsigned)rowStride,                          // tensor_dim0_stride lo
                   (unsigned)((rowStride >> 32) & 0xFFFFu),      // stride0 hi (stride1=0)
                   0u };
  asm volatile("tensor_load_to_lds %0, %1" :: "s"(g0), "s"(g1) : "memory");
}

// ---------------------------------------------------------------------------
// Pointwise (1x1) conv as GEMM:  Y[b, o, n] = sum_c W[o,c] * X[b, c, n]
// Cin == 64 always.  Block = 128 threads (4 waves); block covers a 64-pixel
// strip and up to 128 cout rows (blockIdx.y chunks of 128 for Cout=192).
// TDM stages the W chunk and the 64x64 X tile into LDS once; each wave then
// loops cout tiles with 2 WMMAs each, reusing its B fragments from LDS.
// act: 0 none, 1 exact GELU.  Res (nullable): Y += Res.
// grid: (Npix/64, ceil(Cout/128), B), block 128, dynamic LDS = PW_SHMEM.
// ---------------------------------------------------------------------------
__global__ __launch_bounds__(128) void pw_gemm_wmma(
    const float* __restrict__ X, const float* __restrict__ W,
    float* __restrict__ Y, const float* __restrict__ Res,
    int Cout, int Npix, long xBatchStride, long yBatchStride, int act)
{
  extern __shared__ float smem[];
  float* wAll = smem;                        // [chunkRows][LDS_ROW]
  float* xT   = smem + W_ROWS_MAX * LDS_ROW; // [64][LDS_ROW]

  const int b         = blockIdx.z;
  const int chunkBase = blockIdx.y * W_ROWS_MAX;
  const int chunkRows = min(W_ROWS_MAX, Cout - chunkBase);
  const int lane      = threadIdx.x & 31;
  const int wave      = threadIdx.x >> 5;
  const int half      = lane >> 4;
  const int r         = lane & 15;
  const int n0        = blockIdx.x * 64;
  const int col       = wave * 16 + r;

  const float* Xb = X + (long)b * xBatchStride;
  float*       Yb = Y + (long)b * yBatchStride;

  if (wave == 0) {
    tdm_load_tile_f32(W + (long)chunkBase * 64, wAll, (unsigned)chunkRows, 64, 64ul);
    tdm_load_tile_f32(Xb + n0, xT, 64, 64, (unsigned long)Npix);
    __builtin_amdgcn_s_wait_tensorcnt(0);
  }
  __syncthreads();

  // B fragments for k0 = 0 and k0 = 32, shared across all cout tiles.
  v16h bf0, bf1;
#pragma unroll
  for (int v = 0; v < 8; ++v) {
    const int kk = ((v & 3) * 2) + half * 8 + ((v >> 2) * 16);
    bf0[2 * v]     = (_Float16)xT[kk * LDS_ROW + col];
    bf0[2 * v + 1] = (_Float16)xT[(kk + 1) * LDS_ROW + col];
    bf1[2 * v]     = (_Float16)xT[(kk + 32) * LDS_ROW + col];
    bf1[2 * v + 1] = (_Float16)xT[(kk + 33) * LDS_ROW + col];
  }

  const int tiles = chunkRows >> 4;
  for (int ct = 0; ct < tiles; ++ct) {
    const float* wrow = wAll + (ct * 16 + r) * LDS_ROW;
    v16h a0, a1;
#pragma unroll
    for (int v = 0; v < 8; ++v) {
      const int kk = ((v & 3) * 2) + half * 8 + ((v >> 2) * 16);
      a0[2 * v]     = (_Float16)wrow[kk];
      a0[2 * v + 1] = (_Float16)wrow[kk + 1];
      a1[2 * v]     = (_Float16)wrow[kk + 32];
      a1[2 * v + 1] = (_Float16)wrow[kk + 33];
    }
    v8f acc = {};
    acc = __builtin_amdgcn_wmma_f32_16x16x32_f16(false, a0, false, bf0,
                                                 (short)0, acc, false, false);
    acc = __builtin_amdgcn_wmma_f32_16x16x32_f16(false, a1, false, bf1,
                                                 (short)0, acc, false, false);
#pragma unroll
    for (int rr = 0; rr < 8; ++rr) {
      const int m  = chunkBase + ct * 16 + rr + half * 8;
      const long o = (long)m * Npix + n0 + col;
      float vv = acc[rr];
      if (act == 1) vv = 0.5f * vv * (1.0f + erff(vv * 0.70710678118654752f));
      if (Res) vv += Res[(long)b * yBatchStride + o];
      Yb[o] = vv;
    }
  }
}

// ---------------------------------------------------------------------------
// Depthwise 3x3, pad 1, optional residual.  grid: (H*W/256, C, B), block 256.
// ---------------------------------------------------------------------------
__global__ __launch_bounds__(256) void dwconv3_kernel(
    const float* __restrict__ X, const float* __restrict__ W,
    float* __restrict__ Y, const float* __restrict__ Res,
    int C, int Hh, int Ww)
{
  const int b = blockIdx.z, c = blockIdx.y;
  const long idx = (long)blockIdx.x * 256 + threadIdx.x;
  const int h = (int)(idx / Ww), w = (int)(idx % Ww);
  const float* xp = X + ((long)b * C + c) * (long)Hh * Ww;
  float wc[9];
#pragma unroll
  for (int k = 0; k < 9; ++k) wc[k] = W[c * 9 + k];
  float s = 0.0f;
#pragma unroll
  for (int dy = -1; dy <= 1; ++dy)
#pragma unroll
    for (int dx = -1; dx <= 1; ++dx) {
      const int hh = h + dy, ww = w + dx;
      if (hh >= 0 && hh < Hh && ww >= 0 && ww < Ww)
        s += wc[(dy + 1) * 3 + (dx + 1)] * xp[(long)hh * Ww + ww];
    }
  const long o = ((long)b * C + c) * (long)Hh * Ww + idx;
  if (Res) s += Res[o];
  Y[o] = s;
}

// ---------------------------------------------------------------------------
// Gram of stacked S = [q(8 rows); k(8 rows)] (16 x Npix) per (b,head) using
// WMMA with identical A/B fragments: G = S * S^T.  Diagonal -> ||q||^2,||k||^2,
// block (0..7, 8..15) -> q.k.  Deterministic LDS cross-wave reduction.
// grid: (B*HEADS), block: 256 (8 waves).
// ---------------------------------------------------------------------------
__global__ __launch_bounds__(256) void gram_wmma(
    const float* __restrict__ QKV, float* __restrict__ G, int Npix)
{
  const int bh   = blockIdx.x;
  const int b    = bh >> 3;
  const int hd   = bh & 7;
  const int lane = threadIdx.x & 31;
  const int wave = threadIdx.x >> 5;
  const int half = lane >> 4;
  const int r    = lane & 15;

  // Row r of stacked S: q channels hd*8 + r (r<8), k channels 64 + hd*8 + r-8.
  const int ch = (r < 8) ? (hd * 8 + r) : (64 + hd * 8 + (r - 8));
  const float* Srow = QKV + ((long)b * 192 + ch) * (long)Npix;

  v8f acc = {};
  const int iters = Npix / 256;            // 8 waves x 32 pixels per step
  for (int it = 0; it < iters; ++it) {
    const long p = (long)it * 256 + wave * 32;
    __builtin_prefetch(Srow + p + 256, 0, 3);   // global_prefetch_b8
    v16h s;
#pragma unroll
    for (int v = 0; v < 8; ++v) {
      const long kk = p + ((v & 3) * 2) + half * 8 + ((v >> 2) * 16);
      s[2 * v]     = (_Float16)Srow[kk];
      s[2 * v + 1] = (_Float16)Srow[kk + 1];
    }
    acc = __builtin_amdgcn_wmma_f32_16x16x32_f16(false, s, false, s,
                                                 (short)0, acc, false, false);
  }

  __shared__ float smemg[8 * 256];
#pragma unroll
  for (int rr = 0; rr < 8; ++rr) smemg[wave * 256 + rr * 32 + lane] = acc[rr];
  __syncthreads();
  if (wave == 0) {
#pragma unroll
    for (int rr = 0; rr < 8; ++rr) {
      float s = 0.0f;
      for (int w = 0; w < 8; ++w) s += smemg[w * 256 + rr * 32 + lane];
      const int M = rr + half * 8;
      G[(long)bh * 256 + M * 16 + r] = s;     // D layout: M = rr + half*8, N = r
    }
  }
}

// ---------------------------------------------------------------------------
// Per (b,head): normalize Gram -> attn (8x8), 4 nested top-k softmaxes, mix.
// grid: B*HEADS, block: 64 (one thread per matrix element).
// ---------------------------------------------------------------------------
__global__ __launch_bounds__(64) void sparse_softmax_kernel(
    const float* __restrict__ G, const float* __restrict__ temp,
    const float* __restrict__ mix, float* __restrict__ Spar)
{
  const int bh = blockIdx.x;
  const int hd = bh & 7;
  const int t  = threadIdx.x;
  const int c  = t >> 3, d = t & 7;
  const float* g = G + (long)bh * 256;

  const float nq = fmaxf(sqrtf(fmaxf(g[c * 16 + c], 0.0f)), 1e-12f);
  const float nk = fmaxf(sqrtf(fmaxf(g[(8 + d) * 16 + 8 + d], 0.0f)), 1e-12f);
  const float a  = g[c * 16 + 8 + d] / (nq * nk) * temp[hd];

  __shared__ float vals[64];
  __shared__ float es[64];
  __shared__ int   rk[64];
  vals[t] = a;
  __syncthreads();

  int rank = 0;
  float vmax = -INFINITY;
  for (int j = 0; j < 64; ++j) {
    const float vj = vals[j];
    rank += (vj > a) || (vj == a && j < t);   // top_k ties: lower index first
    vmax = fmaxf(vmax, vj);
  }
  const float e = expf(a - vmax);
  es[t] = e;
  rk[t] = rank;
  __syncthreads();

  const int ks[4] = {32, 42, 48, 51};        // CC*{1/2,2/3,3/4,4/5}, CC=64
  float outv = 0.0f;
#pragma unroll
  for (int i = 0; i < 4; ++i) {
    float sum = 0.0f;
    for (int j = 0; j < 64; ++j) sum += (rk[j] < ks[i]) ? es[j] : 0.0f;
    if (rank < ks[i]) outv += mix[i] * (e / sum);
  }
  Spar[(long)bh * 64 + t] = outv;
}

// ---------------------------------------------------------------------------
// out[b, hd*8+c, n] = sum_d spar[c,d] * v[b, 128+hd*8+d, n]
// grid: (Npix/256, HEADS, B), block 256 (one pixel per thread, all 8 c).
// ---------------------------------------------------------------------------
__global__ __launch_bounds__(256) void attn_apply_kernel(
    const float* __restrict__ Spar, const float* __restrict__ QKV,
    float* __restrict__ Out, int Npix)
{
  const int b = blockIdx.z, hd = blockIdx.y;
  const long n = (long)blockIdx.x * 256 + threadIdx.x;
  const float* vb = QKV + ((long)b * 192 + 128 + hd * 8) * (long)Npix + n;
  float vv[8];
#pragma unroll
  for (int d = 0; d < 8; ++d) vv[d] = vb[(long)d * Npix];
  const float* sp = Spar + ((long)b * HEADS + hd) * 64;
  float* ob = Out + ((long)b * DIM + hd * 8) * (long)Npix + n;
#pragma unroll
  for (int c = 0; c < 8; ++c) {
    float s = 0.0f;
#pragma unroll
    for (int d = 0; d < 8; ++d) s += sp[c * 8 + d] * vv[d];
    ob[(long)c * Npix] = s;
  }
}

// ---------------------------------------------------------------------------
// LayerNorm over channel dim (64).  grid: (Npix/256, B), block 256.
// ---------------------------------------------------------------------------
__global__ __launch_bounds__(256) void ln2d_kernel(
    const float* __restrict__ X, const float* __restrict__ w,
    const float* __restrict__ bias, float* __restrict__ Y, int Npix)
{
  const int b = blockIdx.y;
  const long n = (long)blockIdx.x * 256 + threadIdx.x;
  const float* xp = X + (long)b * DIM * Npix + n;
  float mu = 0.0f;
  for (int c = 0; c < DIM; ++c) mu += xp[(long)c * Npix];
  mu *= (1.0f / DIM);
  float var = 0.0f;
  for (int c = 0; c < DIM; ++c) {
    const float d = xp[(long)c * Npix] - mu;
    var += d * d;
  }
  var *= (1.0f / DIM);
  const float inv = rsqrtf(var + EPS_LN);
  float* yp = Y + (long)b * DIM * Npix + n;
  for (int c = 0; c < DIM; ++c)
    yp[(long)c * Npix] = w[c] * ((xp[(long)c * Npix] - mu) * inv) + bias[c];
}

// ---------------------------------------------------------------------------
// 2x2 average pool.  grid: ((H/2*W/2)/256, 64, B).
// ---------------------------------------------------------------------------
__global__ __launch_bounds__(256) void avgpool2_kernel(
    const float* __restrict__ X, float* __restrict__ Y,
    int Hh, int Ww, long xBatchStride, long yBatchStride)
{
  const int b = blockIdx.z, c = blockIdx.y;
  const int Ho = Hh >> 1, Wo = Ww >> 1;
  const long idx = (long)blockIdx.x * 256 + threadIdx.x;
  const int ho = (int)(idx / Wo), wo = (int)(idx % Wo);
  const float* xp = X + (long)b * xBatchStride + (long)c * Hh * Ww
                      + (long)(2 * ho) * Ww + 2 * wo;
  Y[(long)b * yBatchStride + (long)c * Ho * Wo + idx] =
      0.25f * (xp[0] + xp[1] + xp[Ww] + xp[Ww + 1]);
}

// ---------------------------------------------------------------------------
// prod = x1 * nearest_upsample2(x2s).  grid: (H*W/256, 64, B).
// ---------------------------------------------------------------------------
__global__ __launch_bounds__(256) void mul_up_kernel(
    const float* __restrict__ X1, const float* __restrict__ X2s,
    float* __restrict__ P, int Hh, int Ww)
{
  const int b = blockIdx.z, c = blockIdx.y;
  const int Ho = Hh >> 1, Wo = Ww >> 1;
  const long idx = (long)blockIdx.x * 256 + threadIdx.x;
  const int h = (int)(idx / Ww), w = (int)(idx % Ww);
  const float x2 = X2s[((long)b * DIM + c) * (long)Ho * Wo
                       + (long)(h >> 1) * Wo + (w >> 1)];
  const long o = ((long)b * DIM + c) * (long)Hh * Ww + idx;
  P[o] = X1[o] * x2;
}

// ---------------------------------------------------------------------------
extern "C" void kernel_launch(void* const* d_in, const int* in_sizes, int n_in,
                              void* d_out, int out_size, void* d_ws, size_t ws_size,
                              hipStream_t stream) {
  (void)in_sizes; (void)n_in; (void)ws_size;
  const long N = NPIX;                     // 65536 pixels per batch image
  const long Nq = N / 4;                   // pooled

  float* ws = (float*)d_ws;
  // Workspace layout (floats), liveness-aliased:
  size_t off = 0;
  float* bigA = ws + off; off += (size_t)BATCH * 192 * N;  // qkv_pre, later FF(128ch)
  float* bigB = ws + off; off += (size_t)BATCH * 192 * N;  // qkv (post dwconv)
  float* c0   = ws + off; off += (size_t)BATCH * DIM * N;  // running x
  float* c1   = ws + off; off += (size_t)BATCH * DIM * N;  // attn-out / T1 / prod
  float* c2   = ws + off; off += (size_t)BATCH * DIM * N;  // x after attn residual
  float* c3   = ws + off; off += (size_t)BATCH * DIM * N;  // LN / x1 / T3
  float* p0   = ws + off; off += (size_t)BATCH * DIM * Nq; // pooled
  float* p1   = ws + off; off += (size_t)BATCH * DIM * Nq; // g2 pw out
  float* p2   = ws + off; off += (size_t)BATCH * DIM * Nq; // x2 small
  float* gacc = ws + off; off += (size_t)BATCH * HEADS * 256;
  float* spar = ws + off; off += (size_t)BATCH * HEADS * 64;

  // x -> running buffer
  hipMemcpyAsync(c0, d_in[0], (size_t)BATCH * DIM * N * sizeof(float),
                 hipMemcpyDeviceToDevice, stream);

  for (int i = 0; i < NB; ++i) {
    const float* qkv_w = (const float*)d_in[1]  + (size_t)i * 192 * 64;
    const float* dw_w  = (const float*)d_in[2]  + (size_t)i * 192 * 9;
    const float* proj_w= (const float*)d_in[3]  + (size_t)i * 64 * 64;
    const float* temp  = (const float*)d_in[4]  + (size_t)i * HEADS;
    const float* mix   = (const float*)d_in[5]  + (size_t)i * 4;
    const float* ln_w  = (const float*)d_in[6]  + (size_t)i * 64;
    const float* ln_b  = (const float*)d_in[7]  + (size_t)i * 64;
    const float* ff_w  = (const float*)d_in[8]  + (size_t)i * 128 * 64;
    const float* g1_pw = (const float*)d_in[9]  + (size_t)i * 64 * 64;
    const float* g1_dw = (const float*)d_in[10] + (size_t)i * 64 * 9;
    const float* g2_pw = (const float*)d_in[11] + (size_t)i * 64 * 64;
    const float* g2_dw = (const float*)d_in[12] + (size_t)i * 64 * 9;
    const float* g3_pw = (const float*)d_in[13] + (size_t)i * 64 * 64;
    const float* g3_dw = (const float*)d_in[14] + (size_t)i * 64 * 9;

    // ---- ss_sa ----
    // 1) qkv pre: pw 64 -> 192 (two 128/64-row chunks)
    pw_gemm_wmma<<<dim3(N / 64, 2, BATCH), 128, PW_SHMEM, stream>>>(
        c0, qkv_w, bigA, nullptr, 192, (int)N, 64 * N, 192 * N, 0);
    // 2) depthwise 3x3 on 192 channels
    dwconv3_kernel<<<dim3(N / 256, 192, BATCH), 256, 0, stream>>>(
        bigA, dw_w, bigB, nullptr, 192, IMH, IMW);
    // 3) Gram of [q;k] per (b,head) via WMMA
    gram_wmma<<<dim3(BATCH * HEADS), 256, 0, stream>>>(bigB, gacc, (int)N);
    // 4) sparse-softmax mixing
    sparse_softmax_kernel<<<dim3(BATCH * HEADS), 64, 0, stream>>>(
        gacc, temp, mix, spar);
    // 5) spar @ v
    attn_apply_kernel<<<dim3(N / 256, HEADS, BATCH), 256, 0, stream>>>(
        spar, bigB, c1, (int)N);
    // 6) proj pw 64->64 with residual (+x) -> c2 is new x
    pw_gemm_wmma<<<dim3(N / 64, 1, BATCH), 128, PW_SHMEM, stream>>>(
        c1, proj_w, c2, c0, 64, (int)N, 64 * N, 64 * N, 0);

    // ---- clfn(ln2d(x)) + x ----
    // 7) LayerNorm2d -> c3
    ln2d_kernel<<<dim3(N / 256, BATCH), 256, 0, stream>>>(c2, ln_w, ln_b, c3, (int)N);
    // 8) ff pw 64 -> 128 with exact GELU -> bigA (128ch)
    pw_gemm_wmma<<<dim3(N / 64, 1, BATCH), 128, PW_SHMEM, stream>>>(
        c3, ff_w, bigA, nullptr, 128, (int)N, 64 * N, 128 * N, 1);
    // 9) x1 branch: pw on channels 0..63 -> c1, dwconv -> c3
    pw_gemm_wmma<<<dim3(N / 64, 1, BATCH), 128, PW_SHMEM, stream>>>(
        bigA, g1_pw, c1, nullptr, 64, (int)N, 128 * N, 64 * N, 0);
    dwconv3_kernel<<<dim3(N / 256, 64, BATCH), 256, 0, stream>>>(
        c1, g1_dw, c3, nullptr, 64, IMH, IMW);
    // 10) x2 branch: avgpool channels 64..127 -> p0, pw -> p1, dwconv -> p2
    avgpool2_kernel<<<dim3(Nq / 256, 64, BATCH), 256, 0, stream>>>(
        bigA + 64 * N, p0, IMH, IMW, 128 * N, 64 * Nq);
    pw_gemm_wmma<<<dim3(Nq / 64, 1, BATCH), 128, PW_SHMEM, stream>>>(
        p0, g2_pw, p1, nullptr, 64, (int)Nq, 64 * Nq, 64 * Nq, 0);
    dwconv3_kernel<<<dim3(Nq / 256, 64, BATCH), 256, 0, stream>>>(
        p1, g2_dw, p2, nullptr, 64, IMH / 2, IMW / 2);
    // 11) gate: prod = x1 * up(x2) -> c1
    mul_up_kernel<<<dim3(N / 256, 64, BATCH), 256, 0, stream>>>(
        c3, p2, c1, IMH, IMW);
    // 12) g3 pw -> c3, dwconv with residual (+c2) -> c0 (new running x)
    pw_gemm_wmma<<<dim3(N / 64, 1, BATCH), 128, PW_SHMEM, stream>>>(
        c1, g3_pw, c3, nullptr, 64, (int)N, 64 * N, 64 * N, 0);
    dwconv3_kernel<<<dim3(N / 256, 64, BATCH), 256, 0, stream>>>(
        c3, g3_dw, c0, c2, 64, IMH, IMW);
  }

  hipMemcpyAsync(d_out, c0, (size_t)out_size * sizeof(float),
                 hipMemcpyDeviceToDevice, stream);
}
